// rev_gru_layer_53824530153822
// MI455X (gfx1250) — matrix operand order repaired and checked by reading, not compiled, not run on hardware
//
#include <hip/hip_runtime.h>

// Problem constants (B,T,D,H) from reference
static constexpr int B_  = 256;
static constexpr int T_  = 512;
static constexpr int D_  = 512;
static constexpr int H_  = 256;
static constexpr int K3H = 768;   // 3*H
static constexpr int PB  = 264;   // padded LDS row stride (elements) -> bank rotation

typedef __bf16 bf16_t;
typedef __attribute__((ext_vector_type(16))) __bf16 v16bf;
typedef __attribute__((ext_vector_type(8)))  __bf16 v8bf;
typedef __attribute__((ext_vector_type(8)))  float  v8f;

// fp32 -> bf16, round to nearest even (epilogue-only)
__device__ __forceinline__ bf16_t f2bf(float f) {
  unsigned u = __builtin_bit_cast(unsigned, f);
  unsigned r = u + 0x7FFFu + ((u >> 16) & 1u);
  unsigned short h = (unsigned short)(r >> 16);
  return __builtin_bit_cast(bf16_t, h);
}

__device__ __forceinline__ v8f bc8(float v) {
  v8f a = {v, v, v, v, v, v, v, v};
  return a;
}

__device__ __forceinline__ float sigmoidf_(float x) {
  return 1.0f / (1.0f + __expf(-x));
}

// ---------------------------------------------------------------------------
// A-fragment (16x32 MxK, bf16) from an LDS tile [16][PB].
// lane<16: row=lane, K {0..15}->v[0..15]
// lane>=16: row=lane-16, K {16..31}
// ---------------------------------------------------------------------------
__device__ __forceinline__ v16bf loadA_lds(const bf16_t* __restrict__ Abase, int k0) {
  int lane = threadIdx.x & 31;
  int half = lane >> 4;
  const bf16_t* p = Abase + (lane & 15) * PB + k0 + half * 16;
  v8bf lo = *reinterpret_cast<const v8bf*>(p);
  v8bf hi = *reinterpret_cast<const v8bf*>(p + 8);
  return __builtin_shufflevector(lo, hi, 0, 1, 2, 3, 4, 5, 6, 7,
                                 8, 9, 10, 11, 12, 13, 14, 15);
}

// B-fragment (32x16 KxN), B[k,n] = W[n0+n, k0+k], W row-major [Ntot x 256] bf16
// in global (L2-resident). One contiguous 32B chunk of a weight row per lane.
__device__ __forceinline__ v16bf loadB_w(const bf16_t* __restrict__ W,
                                         int n0, int k0) {
  int lane = threadIdx.x & 31;
  const bf16_t* p = W + (long long)(n0 + (lane & 15)) * H_ + k0 + ((lane >> 4) << 4);
  return *reinterpret_cast<const v16bf*>(p);
}

__device__ __forceinline__ v8f wmma_bf(v16bf a, v16bf b, v8f c) {
  return __builtin_amdgcn_wmma_f32_16x16x32_bf16(false, a, false, b, (short)0, c,
                                                 false, false);
}

// GRU gate epilogue for one 16x16 tile: h_new = (1-z)*tanh(gxn + r*ghn) + z*h
__device__ __forceinline__ void gate_epi(v8f aR, v8f aZ, v8f aXN, v8f aHN,
                                         float* __restrict__ hfp,
                                         bf16_t* __restrict__ hbf, int col) {
  int lane = threadIdx.x & 31;
  int rbase = (lane >> 4) << 3;
#pragma unroll
  for (int i = 0; i < 8; ++i) {
    int row = rbase + i;
    float r  = sigmoidf_(aR[i]);
    float z  = sigmoidf_(aZ[i]);
    float nn = tanhf(aXN[i] + r * aHN[i]);
    float hp = hfp[row * PB + col];
    float hv = (1.0f - z) * nn + z * hp;
    hfp[row * PB + col] = hv;
    hbf[row * PB + col] = f2bf(hv);
  }
}

// ---------------------------------------------------------------------------
// One-time: weights fp32 -> bf16 (then L2-resident for all 512 steps)
// ---------------------------------------------------------------------------
__global__ void __launch_bounds__(256) cvt_weights_kernel(
    const float* __restrict__ Wih, const float* __restrict__ Whh,
    const float* __restrict__ Wout,
    bf16_t* __restrict__ WihBf, bf16_t* __restrict__ WhhBf,
    bf16_t* __restrict__ WoutBf) {
  int i = blockIdx.x * 256 + threadIdx.x;
  const int nih  = 4 * K3H * H_;  // 786432
  const int nout = 4 * H_ * H_;   // 262144
  if (i < nih) { WihBf[i] = f2bf(Wih[i]); WhhBf[i] = f2bf(Whh[i]); }
  if (i < nout) { WoutBf[i] = f2bf(Wout[i]); }
}

// ---------------------------------------------------------------------------
// Persistent kernel: one WG per 16-row batch block, all T steps, state in LDS.
// blockDim = 512 (16 waves). K-loops deliberately NOT unrolled so weight
// loads stay inside the loop (addresses depend on runtime k0) -> no LICM
// hoisting across the t-loop, no VGPR spills; weights stream from L2.
// ---------------------------------------------------------------------------
__global__ void __launch_bounds__(512) rev_gru_persistent_kernel(
    const float* __restrict__ x, float* __restrict__ out,
    const bf16_t* __restrict__ WihBf, const bf16_t* __restrict__ WhhBf,
    const bf16_t* __restrict__ WoutBf,
    const float* __restrict__ bih, const float* __restrict__ bhh,
    const float* __restrict__ bout) {
  extern __shared__ char smem_raw[];
  float*  sh_hfp = (float*)smem_raw;               // 4 subnets x [16][PB] fp32
  bf16_t* sh_hbf = (bf16_t*)(sh_hfp + 4 * 16 * PB);// 4 subnets x [16][PB] bf16
  bf16_t* sh_x2  = sh_hbf + 4 * 16 * PB;           // [16][PB] bf16
  bf16_t* sh_y1  = sh_x2 + 16 * PB;                // [16][PB] bf16

  const int tid  = threadIdx.x;
  const int wv   = tid >> 5;
  const int lane = tid & 31;
  const int c    = lane & 15;
  const int m0   = blockIdx.x * 16;                // global row base
  const long long xrs = (long long)T_ * D_;

  // ---- prologue: zero h state, stage bf16 x2[t=0] ----
  for (int i = tid; i < 4 * 16 * PB; i += 512) sh_hfp[i] = 0.0f;
  const bf16_t zbf = __builtin_bit_cast(bf16_t, (unsigned short)0);
  for (int i = tid; i < 4 * 16 * PB; i += 512) sh_hbf[i] = zbf;
  for (int i = tid; i < 16 * H_; i += 512) {
    int r = i >> 8, cc = i & (H_ - 1);
    sh_x2[r * PB + cc] = f2bf(x[(long long)(m0 + r) * xrs + H_ + cc]);
  }
  __syncthreads();

  // ---- static per-wave assignments (same every step) ----
  const int sA = 2 + (wv >> 3);          // phase A subnet (2 or 3)
  const int sC = (wv >> 3);              // phase C subnet (0 or 1)
  const int jt0 = (wv & 7) * 2;          // first of two jt tiles (gates)
  const int jtB = wv;                    // couple phases: one jt per wave

  const bf16_t* WihA = WihBf + (long long)sA * K3H * H_;
  const bf16_t* WhhA = WhhBf + (long long)sA * K3H * H_;
  const bf16_t* WihC = WihBf + (long long)sC * K3H * H_;
  const bf16_t* WhhC = WhhBf + (long long)sC * K3H * H_;
  float*  hfpA = sh_hfp + sA * 16 * PB;
  bf16_t* hbfA = sh_hbf + sA * 16 * PB;
  float*  hfpC = sh_hfp + sC * 16 * PB;
  bf16_t* hbfC = sh_hbf + sC * 16 * PB;

  // hoisted biases (gate sections r/z/n at col n, H+n, 2H+n)
  const float* bihA = bih + sA * K3H;  const float* bhhA = bhh + sA * K3H;
  const float* bihC = bih + sC * K3H;  const float* bhhC = bhh + sC * K3H;
  const int nA0 = jt0 * 16 + c, nA1 = nA0 + 16;
  const float brA0 = bihA[nA0] + bhhA[nA0],          brA1 = bihA[nA1] + bhhA[nA1];
  const float bzA0 = bihA[H_ + nA0] + bhhA[H_ + nA0], bzA1 = bihA[H_ + nA1] + bhhA[H_ + nA1];
  const float bxA0 = bihA[2 * H_ + nA0],             bxA1 = bihA[2 * H_ + nA1];
  const float bhA0 = bhhA[2 * H_ + nA0],             bhA1 = bhhA[2 * H_ + nA1];
  const float brC0 = bihC[nA0] + bhhC[nA0],          brC1 = bihC[nA1] + bhhC[nA1];
  const float bzC0 = bihC[H_ + nA0] + bhhC[H_ + nA0], bzC1 = bihC[H_ + nA1] + bhhC[H_ + nA1];
  const float bxC0 = bihC[2 * H_ + nA0],             bxC1 = bihC[2 * H_ + nA1];
  const float bhC0 = bhhC[2 * H_ + nA0],             bhC1 = bhhC[2 * H_ + nA1];
  const int nB = jtB * 16 + c;
  const float bsB = bout[2 * H_ + nB], btB = bout[3 * H_ + nB];  // subnets 2,3
  const float bsD = bout[0 * H_ + nB], btD = bout[1 * H_ + nB];  // subnets 0,1

#pragma unroll 1
  for (int t = 0; t < T_; ++t) {
    // ================= Phase A: gates, subnets 2,3, input x2 ================
    {
      v8f r0 = bc8(brA0), z0 = bc8(bzA0), xn0 = bc8(bxA0), hn0 = bc8(bhA0);
      v8f r1 = bc8(brA1), z1 = bc8(bzA1), xn1 = bc8(bxA1), hn1 = bc8(bhA1);
#pragma unroll 1
      for (int k0 = 0; k0 < H_; k0 += 32) {
        v16bf aX = loadA_lds(sh_x2, k0);
        v16bf aH = loadA_lds(hbfA, k0);
        r0  = wmma_bf(aX, loadB_w(WihA, jt0 * 16,            k0), r0);
        r0  = wmma_bf(aH, loadB_w(WhhA, jt0 * 16,            k0), r0);
        z0  = wmma_bf(aX, loadB_w(WihA, H_ + jt0 * 16,       k0), z0);
        z0  = wmma_bf(aH, loadB_w(WhhA, H_ + jt0 * 16,       k0), z0);
        xn0 = wmma_bf(aX, loadB_w(WihA, 2 * H_ + jt0 * 16,   k0), xn0);
        hn0 = wmma_bf(aH, loadB_w(WhhA, 2 * H_ + jt0 * 16,   k0), hn0);
        r1  = wmma_bf(aX, loadB_w(WihA, jt0 * 16 + 16,       k0), r1);
        r1  = wmma_bf(aH, loadB_w(WhhA, jt0 * 16 + 16,       k0), r1);
        z1  = wmma_bf(aX, loadB_w(WihA, H_ + jt0 * 16 + 16,  k0), z1);
        z1  = wmma_bf(aH, loadB_w(WhhA, H_ + jt0 * 16 + 16,  k0), z1);
        xn1 = wmma_bf(aX, loadB_w(WihA, 2 * H_ + jt0 * 16 + 16, k0), xn1);
        hn1 = wmma_bf(aH, loadB_w(WhhA, 2 * H_ + jt0 * 16 + 16, k0), hn1);
      }
      __syncthreads();                       // all gh/x2 reads done
      gate_epi(r0, z0, xn0, hn0, hfpA, hbfA, jt0 * 16 + c);
      gate_epi(r1, z1, xn1, hn1, hfpA, hbfA, jt0 * 16 + 16 + c);
    }
    __syncthreads();                         // h[2],h[3] updated

    // ===== Phase B: out proj subnets 2,3 ; y1 = e(s)*x1 + t ================
    {
      v8f accS = bc8(bsB), accT = bc8(btB);
      const bf16_t* hS = sh_hbf + 2 * 16 * PB;
      const bf16_t* hT = sh_hbf + 3 * 16 * PB;
      const bf16_t* WoS = WoutBf + 2 * H_ * H_;
      const bf16_t* WoT = WoutBf + 3 * H_ * H_;
#pragma unroll 1
      for (int k0 = 0; k0 < H_; k0 += 32) {
        accS = wmma_bf(loadA_lds(hS, k0), loadB_w(WoS, jtB * 16, k0), accS);
        accT = wmma_bf(loadA_lds(hT, k0), loadB_w(WoT, jtB * 16, k0), accT);
      }
      int rbase = (lane >> 4) << 3;
#pragma unroll
      for (int i = 0; i < 8; ++i) {
        int row = rbase + i;
        float s = accS[i], tv = accT[i];
        float e = __expf(3.18f * atanf(s));  // CLAMP*0.636
        long long g = (long long)(m0 + row) * xrs + (long long)t * D_ + nB;
        float y = e * x[g] + tv;             // x1 slice
        out[g] = y;
        sh_y1[row * PB + nB] = f2bf(y);
      }
    }
    __syncthreads();                         // y1 ready

    // ================= Phase C: gates, subnets 0,1, input y1 ================
    {
      v8f r0 = bc8(brC0), z0 = bc8(bzC0), xn0 = bc8(bxC0), hn0 = bc8(bhC0);
      v8f r1 = bc8(brC1), z1 = bc8(bzC1), xn1 = bc8(bxC1), hn1 = bc8(bhC1);
#pragma unroll 1
      for (int k0 = 0; k0 < H_; k0 += 32) {
        v16bf aX = loadA_lds(sh_y1, k0);
        v16bf aH = loadA_lds(hbfC, k0);
        r0  = wmma_bf(aX, loadB_w(WihC, jt0 * 16,            k0), r0);
        r0  = wmma_bf(aH, loadB_w(WhhC, jt0 * 16,            k0), r0);
        z0  = wmma_bf(aX, loadB_w(WihC, H_ + jt0 * 16,       k0), z0);
        z0  = wmma_bf(aH, loadB_w(WhhC, H_ + jt0 * 16,       k0), z0);
        xn0 = wmma_bf(aX, loadB_w(WihC, 2 * H_ + jt0 * 16,   k0), xn0);
        hn0 = wmma_bf(aH, loadB_w(WhhC, 2 * H_ + jt0 * 16,   k0), hn0);
        r1  = wmma_bf(aX, loadB_w(WihC, jt0 * 16 + 16,       k0), r1);
        r1  = wmma_bf(aH, loadB_w(WhhC, jt0 * 16 + 16,       k0), r1);
        z1  = wmma_bf(aX, loadB_w(WihC, H_ + jt0 * 16 + 16,  k0), z1);
        z1  = wmma_bf(aH, loadB_w(WhhC, H_ + jt0 * 16 + 16,  k0), z1);
        xn1 = wmma_bf(aX, loadB_w(WihC, 2 * H_ + jt0 * 16 + 16, k0), xn1);
        hn1 = wmma_bf(aH, loadB_w(WhhC, 2 * H_ + jt0 * 16 + 16, k0), hn1);
      }
      __syncthreads();
      gate_epi(r0, z0, xn0, hn0, hfpC, hbfC, jt0 * 16 + c);
      gate_epi(r1, z1, xn1, hn1, hfpC, hbfC, jt0 * 16 + 16 + c);
    }
    __syncthreads();                         // h[0],h[1] updated

    // ===== Phase D: out proj subnets 0,1 ; y2 = e(s)*x2 + t ; stage x2[t+1] =
    {
      v8f accS = bc8(bsD), accT = bc8(btD);
      const bf16_t* hS = sh_hbf + 0 * 16 * PB;
      const bf16_t* hT = sh_hbf + 1 * 16 * PB;
      const bf16_t* WoS = WoutBf;
      const bf16_t* WoT = WoutBf + H_ * H_;
#pragma unroll 1
      for (int k0 = 0; k0 < H_; k0 += 32) {
        accS = wmma_bf(loadA_lds(hS, k0), loadB_w(WoS, jtB * 16, k0), accS);
        accT = wmma_bf(loadA_lds(hT, k0), loadB_w(WoT, jtB * 16, k0), accT);
      }
      int rbase = (lane >> 4) << 3;
      bool more = (t + 1) < T_;
#pragma unroll
      for (int i = 0; i < 8; ++i) {
        int row = rbase + i;
        float s = accS[i], tv = accT[i];
        float e = __expf(3.18f * atanf(s));
        long long g = (long long)(m0 + row) * xrs + (long long)t * D_ + H_ + nB;
        float y = e * x[g] + tv;             // x2 slice
        out[g] = y;
        if (more) sh_x2[row * PB + nB] = f2bf(x[g + D_]);  // x[:,t+1,H+nB]
      }
    }
    __syncthreads();                         // x2 staged for next step
  }
}

// ---------------------------------------------------------------------------
extern "C" void kernel_launch(void* const* d_in, const int* in_sizes, int n_in,
                              void* d_out, int out_size, void* d_ws, size_t ws_size,
                              hipStream_t stream) {
  const float* x    = (const float*)d_in[0];
  const float* Wih  = (const float*)d_in[1];
  const float* Whh  = (const float*)d_in[2];
  const float* bih  = (const float*)d_in[3];
  const float* bhh  = (const float*)d_in[4];
  const float* Wout = (const float*)d_in[5];
  const float* bout = (const float*)d_in[6];
  float* out = (float*)d_out;

  // workspace: bf16 weights only (~3.6 MB, L2-resident)
  bf16_t* WihBf  = (bf16_t*)d_ws;           // 4*768*256
  bf16_t* WhhBf  = WihBf + 4 * K3H * H_;    // 4*768*256
  bf16_t* WoutBf = WhhBf + 4 * K3H * H_;    // 4*256*256

  cvt_weights_kernel<<<(4 * K3H * H_ + 255) / 256, 256, 0, stream>>>(
      Wih, Whh, Wout, WihBf, WhhBf, WoutBf);

  // LDS: 4x[16][PB] f32 h + 4x[16][PB] bf16 h + x2 + y1 tiles
  size_t smem = (size_t)(4 * 16 * PB) * sizeof(float) +
                (size_t)(6 * 16 * PB) * sizeof(bf16_t);   // 118,272 B
  rev_gru_persistent_kernel<<<B_ / 16, 512, smem, stream>>>(
      x, out, WihBf, WhhBf, WoutBf, bih, bhh, bout);
}
